// FeaTrans_55585466744856
// MI455X (gfx1250) — matrix operand order repaired and checked by reading, not compiled, not run on hardware
//
#include <hip/hip_runtime.h>

// ---------------------------------------------------------------------------
// CDNA5 (gfx1250) bf16-WMMA implementation of the bipartite graph-attention
// reference. All GEMM work routed through v_wmma_f32_16x16x32_bf16.
// Roofline notes (MI455X): adj mask stream = 256 MB (NT loads, read once,
// both heads co-scheduled in one WGP so HBM sees it once ~11us @23.3TB/s);
// attention math ~69 GFLOP on the bf16 WMMA pipes; Q/K/V (12 MB bf16) stay
// resident in the 192 MB L2.
// ---------------------------------------------------------------------------

typedef __bf16 bf16;
typedef __attribute__((ext_vector_type(16))) __bf16 v16bf;
typedef __attribute__((ext_vector_type(8)))  __bf16 v8bf;
typedef __attribute__((ext_vector_type(8)))  float  v8f;

#define NNODES 8192
#define DMODEL 256
#define HEADS  2
#define DH     128
#define SCALE_Q 0.088388347648318447f   // 128^-0.5
#define NEG_BIG (-9.0e15f)

// ---- WMMA fragment load helpers (layouts per CDNA5 ISA 7.12.2) -------------
// A (16x32 bf16): lane m = L&15; koff = (L>=16)?8:0;
//   elems 0-7  = K[koff+0 .. koff+7], elems 8-15 = K[koff+16 .. koff+23]
// caller passes p already offset by koff; both 16B chunks are 16B aligned.
__device__ __forceinline__ v16bf load_frag_a(const bf16* p) {
    v8bf lo = *(const v8bf*)(p);
    v8bf hi = *(const v8bf*)(p + 16);
    return __builtin_shufflevector(lo, hi, 0,1,2,3,4,5,6,7,8,9,10,11,12,13,14,15);
}
// B (32x16 bf16): lane n = L&15; koff = (L>=16)?8:0;
//   elems 0-7 = B[koff+0..7][n], elems 8-15 = B[koff+16..23][n]
// caller passes p = column n of B^T offset by koff.
__device__ __forceinline__ v16bf load_frag_b(const bf16* p) {
    v8bf lo = *(const v8bf*)(p);
    v8bf hi = *(const v8bf*)(p + 16);
    return __builtin_shufflevector(lo, hi, 0,1,2,3,4,5,6,7,8,9,10,11,12,13,14,15);
}
__device__ __forceinline__ v8f wmma_bf16(v16bf a, v16bf b, v8f c) {
    return __builtin_amdgcn_wmma_f32_16x16x32_bf16(false, a, false, b, (short)0, c,
                                                   false, false);
}

// ---------------------------------------------------------------------------
// Kernel 0a: x (fp32) -> xb (bf16), 2M elements, 4 per thread
// ---------------------------------------------------------------------------
__global__ void cvt_x_kernel(const float* __restrict__ x, bf16* __restrict__ xb) {
    int i = (blockIdx.x * blockDim.x + threadIdx.x) * 4;
    float4 v = *(const float4*)(x + i);
    xb[i + 0] = (bf16)v.x;
    xb[i + 1] = (bf16)v.y;
    xb[i + 2] = (bf16)v.z;
    xb[i + 3] = (bf16)v.w;
}

// ---------------------------------------------------------------------------
// Kernel 0b: transpose+convert the five 256x256 fp32 weights to bf16
// (row-major (k,n) -> row-major (n,k)) so B fragments load contiguously.
// ---------------------------------------------------------------------------
__global__ void wtrans_kernel(const float* __restrict__ Wq, const float* __restrict__ Wk,
                              const float* __restrict__ Wv, const float* __restrict__ Wr,
                              const float* __restrict__ Wd,
                              bf16* __restrict__ WqT, bf16* __restrict__ WkT,
                              bf16* __restrict__ WvT, bf16* __restrict__ WrT,
                              bf16* __restrict__ WdT) {
    const int z = blockIdx.z;
    const float* src = (z == 0) ? Wq : (z == 1) ? Wk : (z == 2) ? Wv : (z == 3) ? Wr : Wd;
    bf16* dst = (z == 0) ? WqT : (z == 1) ? WkT : (z == 2) ? WvT : (z == 3) ? WrT : WdT;
    int k = blockIdx.y * 16 + threadIdx.y;
    int n = blockIdx.x * 16 + threadIdx.x;
    dst[n * DMODEL + k] = (bf16)src[k * DMODEL + n];
}

// ---------------------------------------------------------------------------
// Kernel 1: QKV projection.  grid (512, 4, 3), block 128 (4 waves).
// One wave per 16x16 output tile, K=256 in eight k=32 WMMA steps.
// z=0 -> Qb (pre-scaled), z=1 -> Kb, z=2 -> VbT (stored transposed dv x N).
// ---------------------------------------------------------------------------
__global__ void proj_kernel(const bf16* __restrict__ xb,
                            const bf16* __restrict__ WqT, const bf16* __restrict__ WkT,
                            const bf16* __restrict__ WvT,
                            bf16* __restrict__ Qb, bf16* __restrict__ Kb,
                            bf16* __restrict__ VbT) {
    const int lane = threadIdx.x & 31;
    const int wid  = threadIdx.x >> 5;
    const int rt   = blockIdx.x;            // row tile 0..511
    const int ct   = blockIdx.y * 4 + wid;  // col tile 0..15
    const int z    = blockIdx.z;
    const bf16* WT = (z == 0) ? WqT : (z == 1) ? WkT : WvT;

    const int n     = lane & 15;
    const int hi    = lane >> 4;
    const int koff  = hi * 8;
    const int kbase = hi * 16;
    const int arow  = rt * 16 + n;          // A-fragment row (lane m == n)
    const int bcol  = ct * 16 + n;          // B-fragment column

    v8f acc = {};
    #pragma unroll
    for (int kc = 0; kc < DMODEL; kc += 32) {
        v16bf a = load_frag_a(xb + arow * DMODEL + kc + koff);
        v16bf b = load_frag_b(WT + bcol * DMODEL + kc + koff);
        acc = wmma_bf16(a, b, acc);
    }
    #pragma unroll
    for (int r = 0; r < 8; ++r) {
        int gm = rt * 16 + r + hi * 8;      // C layout: VGPR r, lanes 16-31 -> M+8
        int gn = ct * 16 + n;
        if (z == 0)      Qb[gm * DMODEL + gn] = (bf16)(acc[r] * SCALE_Q);
        else if (z == 1) Kb[gm * DMODEL + gn] = (bf16)acc[r];
        else             VbT[gn * NNODES + gm] = (bf16)acc[r];
    }
}

// ---------------------------------------------------------------------------
// Kernel 2: flash-attention. grid (128), block 256 (8 waves):
// waves = 4 row-tiles x 2 heads, so BOTH heads of a row-tile live in the same
// WGP and the 256 MB adj stream is read from HBM only once (NT loads keep it
// from evicting the hot K/V working set in L2).
// Each wave owns 16 query rows for one head, streams all 8192 keys in chunks
// of 32: 8 WMMAs for S = Q K^T, adj masking, online softmax (shfl_xor row
// reduce over the 16-lane halves of the C layout), P staged through LDS
// (C-layout -> A-layout transpose), 8 WMMAs for O += P V.
// ---------------------------------------------------------------------------
__global__ void attn_kernel(const bf16* __restrict__ Qb, const bf16* __restrict__ Kb,
                            const bf16* __restrict__ VbT, const int* __restrict__ adj,
                            bf16* __restrict__ satb) {
    __shared__ __align__(16) bf16 lds_p[8][16 * 32];

    const int lane = threadIdx.x & 31;
    const int wid  = threadIdx.x >> 5;      // 0..7
    const int rt   = blockIdx.x * 4 + (wid & 3);  // query tile 0..511
    const int h    = wid >> 2;              // head 0..1
    const int row0 = rt * 16;
    const int hc   = h * DH;                // head column offset in 256-wide QKV
    const int n     = lane & 15;
    const int hi    = lane >> 4;
    const int koff  = hi * 8;
    const int kbase = hi * 16;

    // Q A-fragments for all four 32-wide dh chunks (Q already scaled)
    v16bf qa[4];
    #pragma unroll
    for (int t = 0; t < 4; ++t)
        qa[t] = load_frag_a(Qb + (row0 + n) * DMODEL + hc + t * 32 + koff);

    v8f O[8];
    #pragma unroll
    for (int t = 0; t < 8; ++t) O[t] = (v8f){};
    float mrow[8], lrow[8];
    #pragma unroll
    for (int r = 0; r < 8; ++r) { mrow[r] = -3.0e38f; lrow[r] = 0.0f; }

    for (int kb = 0; kb < NNODES; kb += 32) {
        // ---- S = Q K^T for keys [kb, kb+32) : two 16x16 tiles ----
        v8f S0 = {}, S1 = {};
        #pragma unroll
        for (int t = 0; t < 4; ++t) {
            v16bf b0 = load_frag_b(Kb + (kb      + n) * DMODEL + hc + t * 32 + koff);
            v16bf b1 = load_frag_b(Kb + (kb + 16 + n) * DMODEL + hc + t * 32 + koff);
            S0 = wmma_bf16(qa[t], b0, S0);
            S1 = wmma_bf16(qa[t], b1, S1);
        }
        // ---- adjacency mask (non-temporal: 256 MB stream, read once) ----
        float P0[8], P1[8];
        #pragma unroll
        for (int r = 0; r < 8; ++r) {
            const int gm = row0 + r + hi * 8;
            const int* arow = adj + (long)gm * NNODES + kb + n;
            int a0  = __builtin_nontemporal_load(arow);
            int a16 = __builtin_nontemporal_load(arow + 16);
            P0[r] = (a0  > 0) ? S0[r] : NEG_BIG;
            P1[r] = (a16 > 0) ? S1[r] : NEG_BIG;
        }
        if (kb + 32 < NNODES)   // stream the adj mask ahead of use
            __builtin_prefetch(adj + (long)(row0 + n) * NNODES + kb + 32, 0, 1);

        // ---- online softmax: row reduce across the 16-lane halves ----
        #pragma unroll
        for (int r = 0; r < 8; ++r) {
            float mx = fmaxf(P0[r], P1[r]);
            mx = fmaxf(mx, __shfl_xor(mx, 1, 32));
            mx = fmaxf(mx, __shfl_xor(mx, 2, 32));
            mx = fmaxf(mx, __shfl_xor(mx, 4, 32));
            mx = fmaxf(mx, __shfl_xor(mx, 8, 32));
            float mn = fmaxf(mrow[r], mx);
            float sc = __expf(mrow[r] - mn);
            mrow[r] = mn;
            lrow[r] *= sc;
            #pragma unroll
            for (int t = 0; t < 8; ++t) O[t][r] *= sc;
            float p0 = __expf(P0[r] - mn);
            float p1 = __expf(P1[r] - mn);
            P0[r] = p0; P1[r] = p1;
            float ps = p0 + p1;
            ps += __shfl_xor(ps, 1, 32);
            ps += __shfl_xor(ps, 2, 32);
            ps += __shfl_xor(ps, 4, 32);
            ps += __shfl_xor(ps, 8, 32);
            lrow[r] += ps;
        }
        // ---- C-layout -> A-layout transpose of P through LDS ----
        bf16* pl = lds_p[wid];
        #pragma unroll
        for (int r = 0; r < 8; ++r) {
            int mm = r + hi * 8;
            pl[mm * 32 + n]      = (bf16)P0[r];
            pl[mm * 32 + 16 + n] = (bf16)P1[r];
        }
        v16bf pa = load_frag_a(pl + n * 32 + koff);
        // ---- O += P V  (V stored transposed: contiguous B fragments) ----
        #pragma unroll
        for (int t = 0; t < 8; ++t) {
            v16bf vb = load_frag_b(VbT + (hc + t * 16 + n) * NNODES + kb + koff);
            O[t] = wmma_bf16(pa, vb, O[t]);
        }
    }
    // ---- finalize: normalize and emit sat (bf16) ----
    #pragma unroll
    for (int r = 0; r < 8; ++r) {
        float inv = 1.0f / lrow[r];
        int gm = row0 + r + hi * 8;
        #pragma unroll
        for (int t = 0; t < 8; ++t)
            satb[gm * DMODEL + hc + t * 16 + n] = (bf16)(O[t][r] * inv);
    }
}

// ---------------------------------------------------------------------------
// Kernel 3: bipartite output projection + bias + leaky-ReLU (fp32 out).
// grid (512, 4), block 128 (4 waves), one 16x16 tile per wave.
// Output written once, never re-read -> non-temporal stores.
// ---------------------------------------------------------------------------
__global__ void outproj_kernel(const bf16* __restrict__ satb,
                               const bf16* __restrict__ WrT, const bf16* __restrict__ WdT,
                               const float* __restrict__ bias, float* __restrict__ out) {
    const int lane = threadIdx.x & 31;
    const int wid  = threadIdx.x >> 5;
    const int rt   = blockIdx.x;            // 0..511 ; rows < 4096 use Wr
    const int ct   = blockIdx.y * 4 + wid;  // 0..15
    const bf16* WT = (rt < 256) ? WrT : WdT;

    const int n     = lane & 15;
    const int hi    = lane >> 4;
    const int koff  = hi * 8;
    const int kbase = hi * 16;

    v8f acc = {};
    #pragma unroll
    for (int kc = 0; kc < DMODEL; kc += 32) {
        v16bf a = load_frag_a(satb + (rt * 16 + n) * DMODEL + kc + koff);
        v16bf b = load_frag_b(WT + (ct * 16 + n) * DMODEL + kc + koff);
        acc = wmma_bf16(a, b, acc);
    }
    #pragma unroll
    for (int r = 0; r < 8; ++r) {
        int gm = rt * 16 + r + hi * 8;
        int gn = ct * 16 + n;
        float o = acc[r] + bias[gn];
        o = (o >= 0.0f) ? o : 0.01f * o;
        __builtin_nontemporal_store(o, out + gm * DMODEL + gn);
    }
}

// ---------------------------------------------------------------------------
extern "C" void kernel_launch(void* const* d_in, const int* in_sizes, int n_in,
                              void* d_out, int out_size, void* d_ws, size_t ws_size,
                              hipStream_t stream) {
    (void)in_sizes; (void)n_in; (void)out_size; (void)ws_size;
    const float* x    = (const float*)d_in[0];
    const int*   adj  = (const int*)d_in[1];
    const float* Wq   = (const float*)d_in[2];
    const float* Wk   = (const float*)d_in[3];
    const float* Wv   = (const float*)d_in[4];
    const float* Wr   = (const float*)d_in[5];
    const float* Wd   = (const float*)d_in[6];
    const float* bias = (const float*)d_in[7];
    float* out = (float*)d_out;

    // workspace carve-up (~20.6 MB, 256B aligned slabs)
    char* p = (char*)d_ws;
    auto take = [&](size_t bytes) { char* q = p; p += (bytes + 255) & ~(size_t)255; return q; };
    bf16* xb   = (bf16*)take((size_t)NNODES * DMODEL * 2);
    bf16* WqT  = (bf16*)take((size_t)DMODEL * DMODEL * 2);
    bf16* WkT  = (bf16*)take((size_t)DMODEL * DMODEL * 2);
    bf16* WvT  = (bf16*)take((size_t)DMODEL * DMODEL * 2);
    bf16* WrT  = (bf16*)take((size_t)DMODEL * DMODEL * 2);
    bf16* WdT  = (bf16*)take((size_t)DMODEL * DMODEL * 2);
    bf16* Qb   = (bf16*)take((size_t)NNODES * DMODEL * 2);
    bf16* Kb   = (bf16*)take((size_t)NNODES * DMODEL * 2);
    bf16* VbT  = (bf16*)take((size_t)NNODES * DMODEL * 2);  // transposed (dv x N)
    bf16* satb = (bf16*)take((size_t)NNODES * DMODEL * 2);

    cvt_x_kernel<<<(NNODES * DMODEL) / (256 * 4), 256, 0, stream>>>(x, xb);
    wtrans_kernel<<<dim3(16, 16, 5), dim3(16, 16), 0, stream>>>(
        Wq, Wk, Wv, Wr, Wd, WqT, WkT, WvT, WrT, WdT);
    proj_kernel<<<dim3(NNODES / 16, 4, 3), 128, 0, stream>>>(
        xb, WqT, WkT, WvT, Qb, Kb, VbT);
    attn_kernel<<<dim3(NNODES / 64), 256, 0, stream>>>(
        Qb, Kb, VbT, adj, satb);
    outproj_kernel<<<dim3(NNODES / 16, 4), 128, 0, stream>>>(
        satb, WrT, WdT, bias, out);
}
